// MoELayer_1769526526370
// MI455X (gfx1250) — matrix operand order repaired and checked by reading, not compiled, not run on hardware
//
#include <hip/hip_runtime.h>
#include <hip/hip_bf16.h>

// ---------------- problem constants (match reference) ----------------
constexpr int NB    = 32768;  // tokens
constexpr int DIN   = 256;
constexpr int DOUTD = 256;
constexpr int NE    = 8;      // experts
constexpr int HID   = 512;
constexpr int NGH   = 128;    // gating hidden
constexpr int BM    = 64;     // tokens per expert-block (4 M-tiles) -> 4 WMMA per B-fragment
constexpr float BAL_COEF = 0.01f;

// dynamic LDS for expert kernel: tok(256B)+wt(256B)+xs(32KB)+hs(64KB)
constexpr size_t EXP_SMEM = 512 + (size_t)BM * DIN * 2 + (size_t)BM * HID * 2;

// ---------------- WMMA fragment types (wave32, 16x16x32 f16) ----------------
typedef __attribute__((ext_vector_type(16))) _Float16 v16h;
typedef __attribute__((ext_vector_type(8)))  _Float16 v8h;
typedef __attribute__((ext_vector_type(4)))  _Float16 v4h;
typedef __attribute__((ext_vector_type(8)))  float    v8f;

static __device__ __forceinline__ v16h make16(v8h lo, v8h hi) {
  return __builtin_shufflevector(lo, hi, 0,1,2,3,4,5,6,7,8,9,10,11,12,13,14,15);
}
static __device__ __forceinline__ v8f wmma16(v16h a, v16h b, v8f c) {
  // D = A(16x32 f16) x B(32x16 f16) + C(16x16 f32)
  return __builtin_amdgcn_wmma_f32_16x16x32_f16(false, a, false, b, (short)0, c, false, false);
}
// A fragment at pre-offset pointer p = base + row*ld + 8*half: halves 0..7 at p, 8..15 at p+16
static __device__ __forceinline__ v16h a_frag_at(const _Float16* p) {
  return make16(*(const v8h*)p, *(const v8h*)(p + 16));
}
// B fragment at pre-offset pointer p = Wt_row + 8*half: halves 0..7 at p, 8..15 at p+16
static __device__ __forceinline__ v16h b_frag_at(const _Float16* p) {
  return make16(*(const v8h*)p, *(const v8h*)(p + 16));
}

// ---------------- utility kernels ----------------
__global__ void zero_kernel(float4* __restrict__ out4, int n4, int* __restrict__ counts) {
  int id = blockIdx.x * blockDim.x + threadIdx.x;
  if (id < n4) out4[id] = make_float4(0.f, 0.f, 0.f, 0.f);
  if (id < NE) counts[id] = 0;
}

__global__ void convert_x_kernel(const float* __restrict__ x, _Float16* __restrict__ xh, int n) {
  int i = (blockIdx.x * blockDim.x + threadIdx.x) * 4;
  if (i < n) {
    float4 v = *(const float4*)(x + i);
    v4h h; h[0] = (_Float16)v.x; h[1] = (_Float16)v.y; h[2] = (_Float16)v.z; h[3] = (_Float16)v.w;
    *(v4h*)(xh + i) = h;
  }
}

// src: [e][Kd][Nd] f32  ->  dst: [e][Nd][Kd] f16 (tiled, conflict-free)
__global__ void transpose_f32_f16(const float* __restrict__ src, _Float16* __restrict__ dst,
                                  int Kd, int Nd) {
  __shared__ float tile[32][33];
  int e = blockIdx.z;
  const float* s = src + (size_t)e * Kd * Nd;
  _Float16*    d = dst + (size_t)e * Kd * Nd;
  int k0 = blockIdx.x * 32, n0 = blockIdx.y * 32;
  int tx = threadIdx.x & 31, ty = threadIdx.x >> 5;  // 256 thr = 8 rows of 32
  #pragma unroll
  for (int i = ty; i < 32; i += 8)
    tile[i][tx] = s[(size_t)(k0 + i) * Nd + n0 + tx];
  __syncthreads();
  #pragma unroll
  for (int i = ty; i < 32; i += 8)
    d[(size_t)(n0 + i) * Kd + k0 + tx] = (_Float16)tile[tx][i];
}

// ---------------- gating: logits = relu(x@gW1+gb1)@gW2+gb2; softmax; top-2; build routes ------
__global__ __launch_bounds__(256) void gating_kernel(
    const _Float16* __restrict__ xh, const _Float16* __restrict__ gW1t,
    const float* __restrict__ gb1, const float* __restrict__ gW2, const float* __restrict__ gb2,
    int* __restrict__ counts, int* __restrict__ idx_list, float* __restrict__ w_list) {
  __shared__ float hbuf[16 * NGH];
  __shared__ float lg[16 * NE];
  int bt0 = blockIdx.x * 16;
  int tid = threadIdx.x;
  int wave = tid >> 5, lane = tid & 31, half = lane >> 4, l16 = lane & 15;

  // layer 1 via WMMA: each of 8 waves owns 16 of the 128 hidden cols
  int n = wave * 16 + l16;
  const _Float16* ap = xh + (size_t)(bt0 + l16) * DIN + 8 * half;
  const _Float16* bp = gW1t + (size_t)n * DIN + 8 * half;
  v8f c = {};
  #pragma unroll
  for (int ks = 0; ks < DIN / 32; ++ks) {
    c = wmma16(a_frag_at(ap + ks * 32), b_frag_at(bp + ks * 32), c);
  }
  float bias = gb1[n];
  #pragma unroll
  for (int v = 0; v < 8; ++v) hbuf[(v + 8 * half) * NGH + n] = fmaxf(c[v] + bias, 0.f);
  __syncthreads();

  // tiny layer 2 (128x8) on VALU
  if (tid < 16 * NE) {
    int r = tid >> 3, e = tid & 7;
    float s = gb2[e];
    #pragma unroll 8
    for (int j = 0; j < NGH; ++j) s += hbuf[r * NGH + j] * gW2[j * NE + e];
    lg[r * NE + e] = s;
  }
  __syncthreads();

  // softmax + top-2 + renorm, emit (token, weight) routes per expert
  if (tid < 16) {
    float p[NE]; float mx = -1e30f;
    #pragma unroll
    for (int e = 0; e < NE; ++e) { p[e] = lg[tid * NE + e]; mx = fmaxf(mx, p[e]); }
    float sum = 0.f;
    #pragma unroll
    for (int e = 0; e < NE; ++e) { p[e] = __expf(p[e] - mx); sum += p[e]; }
    #pragma unroll
    for (int e = 0; e < NE; ++e) p[e] /= sum;
    int i1 = 0;
    #pragma unroll
    for (int e = 1; e < NE; ++e) if (p[e] > p[i1]) i1 = e;   // first max (jax tie-break)
    int i2 = (i1 == 0) ? 1 : 0;
    #pragma unroll
    for (int e = 0; e < NE; ++e) if (e != i1 && p[e] > p[i2]) i2 = e;
    float w1 = p[i1], w2 = p[i2], s2 = w1 + w2;
    w1 /= s2; w2 /= s2;
    int g = bt0 + tid;
    int pos1 = atomicAdd(&counts[i1], 1);
    idx_list[i1 * NB + pos1] = g; w_list[i1 * NB + pos1] = w1;
    int pos2 = atomicAdd(&counts[i2], 1);
    idx_list[i2 * NB + pos2] = g; w_list[i2 * NB + pos2] = w2;
  }
}

// ---------------- routed expert MLP: 64-token tile, 8 waves, 3 WMMA GEMMs, dynamic LDS --------
__global__ __launch_bounds__(256) void expert_kernel(
    const _Float16* __restrict__ xh,
    const _Float16* __restrict__ W1t, const float* __restrict__ be1,
    const _Float16* __restrict__ W2t, const float* __restrict__ be2,
    const _Float16* __restrict__ W3t, const float* __restrict__ be3,
    const int* __restrict__ counts, const int* __restrict__ idx_list,
    const float* __restrict__ w_list, float* __restrict__ out) {
  int e = blockIdx.y;
  int cnt = counts[e];
  int t0 = blockIdx.x * BM;
  if (t0 >= cnt) return;

  extern __shared__ char smem[];
  int*      tok = (int*)smem;                               // 256 B
  float*    wt  = (float*)(smem + 256);                     // 256 B
  _Float16* xs  = (_Float16*)(smem + 512);                  // 32 KB
  _Float16* hs  = (_Float16*)(smem + 512 + BM * DIN * 2);   // 64 KB (h1 then h2)

  int tid = threadIdx.x;
  int wave = tid >> 5, lane = tid & 31, half = lane >> 4, l16 = lane & 15;

  if (tid < BM) {
    int i = t0 + tid;
    if (i < cnt) { tok[tid] = idx_list[e * NB + i]; wt[tid] = w_list[e * NB + i]; }
    else         { tok[tid] = 0;                    wt[tid] = 0.f; }           // padded row -> weight 0
  }
  __syncthreads();

  // gather x rows (f16): 64 rows x 512 B; 256 threads x 128 B
  {
    int r = tid >> 2, cc = (tid & 3) * 64;
    const uint4* s = (const uint4*)(xh + (size_t)tok[r] * DIN + cc);
    uint4* d = (uint4*)(xs + r * DIN + cc);
    #pragma unroll
    for (int j = 0; j < 8; ++j) d[j] = s[j];
  }
  __syncthreads();

  // ---- layer 1: h1 = relu([64x256] @ [256x512] + b1); wave owns 64 cols, 4 M-tiles ----
  {
    const _Float16* W1 = W1t + (size_t)e * HID * DIN;
    const _Float16* apb[4];
    const _Float16* bpb[4];
    #pragma unroll
    for (int mt = 0; mt < 4; ++mt) apb[mt] = xs + (mt * 16 + l16) * DIN + 8 * half;
    #pragma unroll
    for (int nt = 0; nt < 4; ++nt)
      bpb[nt] = W1 + (size_t)(wave * 64 + nt * 16 + l16) * DIN + 8 * half;
    v8f c1[4][4] = {};
    #pragma unroll 2
    for (int ks = 0; ks < DIN / 32; ++ks) {
      int ko = ks * 32;
      v16h a[4];
      #pragma unroll
      for (int mt = 0; mt < 4; ++mt) a[mt] = a_frag_at(apb[mt] + ko);
      #pragma unroll
      for (int nt = 0; nt < 4; ++nt) {
        v16h b = b_frag_at(bpb[nt] + ko);
        #pragma unroll
        for (int mt = 0; mt < 4; ++mt) c1[mt][nt] = wmma16(a[mt], b, c1[mt][nt]);
      }
    }
    #pragma unroll
    for (int nt = 0; nt < 4; ++nt) {
      int n = wave * 64 + nt * 16 + l16;
      float bias = be1[e * HID + n];
      #pragma unroll
      for (int mt = 0; mt < 4; ++mt)
        #pragma unroll
        for (int v = 0; v < 8; ++v)
          hs[(mt * 16 + v + 8 * half) * HID + n] = (_Float16)fmaxf(c1[mt][nt][v] + bias, 0.f);
    }
  }
  __syncthreads();

  // ---- layer 2: h2 = relu([64x512] @ [512x512] + b2); accumulate in regs, then overwrite hs ----
  {
    const _Float16* W2 = W2t + (size_t)e * HID * HID;
    const _Float16* apb[4];
    const _Float16* bpb[4];
    #pragma unroll
    for (int mt = 0; mt < 4; ++mt) apb[mt] = hs + (mt * 16 + l16) * HID + 8 * half;
    #pragma unroll
    for (int nt = 0; nt < 4; ++nt)
      bpb[nt] = W2 + (size_t)(wave * 64 + nt * 16 + l16) * HID + 8 * half;
    v8f c2[4][4] = {};
    #pragma unroll 2
    for (int ks = 0; ks < HID / 32; ++ks) {
      int ko = ks * 32;
      v16h a[4];
      #pragma unroll
      for (int mt = 0; mt < 4; ++mt) a[mt] = a_frag_at(apb[mt] + ko);
      #pragma unroll
      for (int nt = 0; nt < 4; ++nt) {
        v16h b = b_frag_at(bpb[nt] + ko);
        #pragma unroll
        for (int mt = 0; mt < 4; ++mt) c2[mt][nt] = wmma16(a[mt], b, c2[mt][nt]);
      }
    }
    __syncthreads();  // everyone done reading h1
    #pragma unroll
    for (int nt = 0; nt < 4; ++nt) {
      int n = wave * 64 + nt * 16 + l16;
      float bias = be2[e * HID + n];
      #pragma unroll
      for (int mt = 0; mt < 4; ++mt)
        #pragma unroll
        for (int v = 0; v < 8; ++v)
          hs[(mt * 16 + v + 8 * half) * HID + n] = (_Float16)fmaxf(c2[mt][nt][v] + bias, 0.f);
    }
  }
  __syncthreads();

  // ---- layer 3: y = [64x512] @ [512x256] + b3; scale by gate weight, scatter-accumulate ----
  {
    const _Float16* W3 = W3t + (size_t)e * DOUTD * HID;
    const _Float16* apb[4];
    const _Float16* bpb[2];
    #pragma unroll
    for (int mt = 0; mt < 4; ++mt) apb[mt] = hs + (mt * 16 + l16) * HID + 8 * half;
    #pragma unroll
    for (int nt = 0; nt < 2; ++nt)
      bpb[nt] = W3 + (size_t)(wave * 32 + nt * 16 + l16) * HID + 8 * half;
    v8f c3[4][2] = {};
    #pragma unroll 2
    for (int ks = 0; ks < HID / 32; ++ks) {
      int ko = ks * 32;
      v16h a[4];
      #pragma unroll
      for (int mt = 0; mt < 4; ++mt) a[mt] = a_frag_at(apb[mt] + ko);
      #pragma unroll
      for (int nt = 0; nt < 2; ++nt) {
        v16h b = b_frag_at(bpb[nt] + ko);
        #pragma unroll
        for (int mt = 0; mt < 4; ++mt) c3[mt][nt] = wmma16(a[mt], b, c3[mt][nt]);
      }
    }
    #pragma unroll
    for (int nt = 0; nt < 2; ++nt) {
      int n = wave * 32 + nt * 16 + l16;
      float bias = be3[e * DOUTD + n];
      #pragma unroll
      for (int mt = 0; mt < 4; ++mt)
        #pragma unroll
        for (int v = 0; v < 8; ++v) {
          int r = mt * 16 + v + 8 * half;
          float val = (c3[mt][nt][v] + bias) * wt[r];  // exactly 2 adds/element -> deterministic
          unsafeAtomicAdd(&out[(size_t)tok[r] * DOUTD + n], val);
        }
    }
  }
}

// ---------------- aux outputs: usage + balance loss ----------------
__global__ void finalize_kernel(const int* __restrict__ counts, float* __restrict__ out_tail) {
  if (threadIdx.x == 0) {
    float loss = 0.f;
    for (int e = 0; e < NE; ++e) {
      float u = counts[e] * (1.0f / NB);
      out_tail[1 + e] = u;
      float d = u - 1.0f / NE;
      loss += d * d;
    }
    out_tail[0] = loss * (1.0f / NE) * BAL_COEF;
  }
}

// ---------------- host launcher ----------------
extern "C" void kernel_launch(void* const* d_in, const int* in_sizes, int n_in,
                              void* d_out, int out_size, void* d_ws, size_t ws_size,
                              hipStream_t stream) {
  const float* x   = (const float*)d_in[0];
  const float* gW1 = (const float*)d_in[1];
  const float* gb1 = (const float*)d_in[2];
  const float* gW2 = (const float*)d_in[3];
  const float* gb2 = (const float*)d_in[4];
  const float* We1 = (const float*)d_in[5];
  const float* be1 = (const float*)d_in[6];
  const float* We2 = (const float*)d_in[7];
  const float* be2 = (const float*)d_in[8];
  const float* We3 = (const float*)d_in[9];
  const float* be3 = (const float*)d_in[10];
  float* out = (float*)d_out;

  char* ws = (char*)d_ws;
  size_t off = 0;
  auto walloc = [&](size_t bytes) -> char* {
    char* p = ws + off; off += (bytes + 255) & ~(size_t)255; return p;
  };
  _Float16* xh    = (_Float16*)walloc((size_t)NB * DIN * 2);        // 16 MB
  _Float16* gW1t  = (_Float16*)walloc((size_t)NGH * DIN * 2);       // 64 KB
  _Float16* We1t  = (_Float16*)walloc((size_t)NE * HID * DIN * 2);  // 2 MB
  _Float16* We2t  = (_Float16*)walloc((size_t)NE * HID * HID * 2);  // 4 MB
  _Float16* We3t  = (_Float16*)walloc((size_t)NE * DOUTD * HID * 2);// 2 MB
  int*      counts   = (int*)walloc(NE * sizeof(int));
  int*      idx_list = (int*)walloc((size_t)NE * NB * sizeof(int));     // 1 MB
  float*    w_list   = (float*)walloc((size_t)NE * NB * sizeof(float)); // 1 MB

  // allow >64KB dynamic LDS for the expert kernel (gfx1250 WGP has 320KB)
  hipFuncSetAttribute(reinterpret_cast<const void*>(expert_kernel),
                      hipFuncAttributeMaxDynamicSharedMemorySize, (int)EXP_SMEM);

  // 1) zero out-accumulator + route counters
  {
    int n4 = NB * DOUTD / 4;
    zero_kernel<<<(n4 + 255) / 256, 256, 0, stream>>>((float4*)out, n4, counts);
  }
  // 2) f32 -> f16 conversions (x row-major; weights transposed to [N][K])
  convert_x_kernel<<<(NB * DIN / 4 + 255) / 256, 256, 0, stream>>>(x, xh, NB * DIN);
  transpose_f32_f16<<<dim3(DIN / 32, NGH / 32, 1),  256, 0, stream>>>(gW1, gW1t, DIN, NGH);
  transpose_f32_f16<<<dim3(DIN / 32, HID / 32, NE), 256, 0, stream>>>(We1, We1t, DIN, HID);
  transpose_f32_f16<<<dim3(HID / 32, HID / 32, NE), 256, 0, stream>>>(We2, We2t, HID, HID);
  transpose_f32_f16<<<dim3(HID / 32, DOUTD / 32, NE), 256, 0, stream>>>(We3, We3t, HID, DOUTD);
  // 3) gating + top-2 routing
  gating_kernel<<<NB / 16, 256, 0, stream>>>(xh, gW1t, gb1, gW2, gb2, counts, idx_list, w_list);
  // 4) routed expert MLP (blocks beyond each expert's count exit immediately)
  expert_kernel<<<dim3(NB / BM, NE, 1), 256, EXP_SMEM, stream>>>(
      xh, We1t, be1, We2t, be2, We3t, be3, counts, idx_list, w_list, out);
  // 5) usage + balance loss
  finalize_kernel<<<1, 32, 0, stream>>>(counts, out + (size_t)NB * DOUTD);
}
